// FastWaveNet_962072674958
// MI455X (gfx1250) — matrix operand order repaired and checked by reading, not compiled, not run on hardware
//
#include <hip/hip_runtime.h>
#include <hip/hip_bf16.h>

// ---------------------------------------------------------------------------
// FastWaveNet autoregressive sampler for MI455X (gfx1250, wave32).
//
// Design:
//  * Kernel 1 (cond_gemm): condition_W(1280x80) @ y(80x8) via
//    V_WMMA_F32_16X16X32_F16, result (frame-major) into d_ws.
//    K padded 80->96 with compile-time zero fill (no per-element predication).
//  * Kernel 2 (wavenet_scan): ONE persistent workgroup (256 thr = 8 wave32)
//    runs all 512 sequential steps. Dilation history lives entirely in LDS
//    as per-layer ring buffers (1023*64 floats ~ 256KB of the 320KB WGP LDS).
//    Per-layer gate matvec h = [WV_past|WV_pres](128x128) @ [past;x] is done
//    with V_WMMA_F32_16X16X4_F32: 8 waves x 1 row-tile x 32 K-steps,
//    input vector broadcast into all 16 B columns so every D column == h.
// ---------------------------------------------------------------------------

typedef __attribute__((ext_vector_type(2)))  float    v2f;
typedef __attribute__((ext_vector_type(8)))  float    v8f;
typedef __attribute__((ext_vector_type(16))) _Float16 v16h;

#define WN_L     10
#define WN_R     64
#define WN_G     128
#define WN_S     256
#define WN_MEL   80
#define WN_NCLS  256
#define WN_T     512
#define WN_CROWS 1280   // L*G

// ---- LDS layout (floats) for the scan kernel ------------------------------
#define RING_F   65472            // sum_{j=0..9} 2^j * 64 = 1023*64
#define X_OFF    (RING_F)         // x[64]
#define V_OFF    (X_OFF + 64)     // v[128] = concat(past, x)
#define H_OFF    (V_OFF + 128)    // h[128] gate pre-activation (incl. cond)
#define Z_OFF    (H_OFF + 128)    // z[64]
#define SK_OFF   (Z_OFF + 64)     // relu(skip)[256]
#define HID_OFF  (SK_OFF + 256)   // hid[256]
#define SCN_OFF  (HID_OFF + 256)  // prefix-scan buffer[256]
#define RED_OFF  (SCN_OFF + 256)  // max-reduce buffer[256]
#define LDS_F    (RED_OFF + 256)  // = 66880 floats
#define LDS_BYTES (LDS_F * 4 + 32) // + small int scratch, padded

#if defined(__has_builtin)
#  if __has_builtin(__builtin_amdgcn_wmma_f32_16x16x4_f32)
#    define HAVE_WMMA_F32X4 1
#  endif
#endif

// fp32 dot over 64 elements, vectorized global row loads (16B aligned rows)
__device__ __forceinline__ float dot64(const float* __restrict__ row,
                                       const float* __restrict__ z) {
  float acc = 0.f;
#pragma unroll
  for (int k = 0; k < 64; k += 4) {
    float4 w = *(const float4*)(row + k);
    acc += w.x * z[k] + w.y * z[k + 1] + w.z * z[k + 2] + w.w * z[k + 3];
  }
  return acc;
}

__device__ __forceinline__ float dot256(const float* __restrict__ row,
                                        const float* __restrict__ z) {
  float acc = 0.f;
#pragma unroll 8
  for (int k = 0; k < 256; k += 4) {
    float4 w = *(const float4*)(row + k);
    acc += w.x * z[k] + w.y * z[k + 1] + w.z * z[k + 2] + w.w * z[k + 3];
  }
  return acc;
}

// ---------------------------------------------------------------------------
// Kernel 1: cond_frames[f][row] = (condition_W @ y)[row][f]
// One wave per 16-row tile (80 tiles). K=80 padded to 96 (3 x 16x16x32 WMMA).
// Padding is resolved at compile time (kc unrolled), loads are float4.
// ---------------------------------------------------------------------------
__global__ void __launch_bounds__(32, 1)
cond_gemm(const float* __restrict__ CW,   // (1280, 80) row-major
          const float* __restrict__ y,    // (80, 8) row-major
          float* __restrict__ condf) {    // (8, 1280) frame-major
  const int lane = threadIdx.x;           // 0..31 (wave32)
  const int m0   = blockIdx.x * 16;       // tile 0..79
  const int mrow = m0 + (lane & 15);
  const int hiK  = lane >> 4;             // 0 or 1
  const float* __restrict__ arow = CW + mrow * WN_MEL;

  v8f acc = {};
#pragma unroll
  for (int kc = 0; kc < 96; kc += 32) {
    // ---- A fragment: 16x32 f16. ISA layout per lane:
    //   lane<16 : halves 0..7 -> K kc+0..7,  halves 8..15 -> K kc+16..23
    //   lane>=16: halves 0..7 -> K kc+8..15, halves 8..15 -> K kc+24..31
    // i.e. halves 0..7 start at kA0 = kc + hiK*8, halves 8..15 at kA0+16.
    const int kA0 = kc + hiK * 8;       // kc<64: all valid; kc==64: K 64..79
    v16h a;
    {
      float4 w0 = *(const float4*)(arow + kA0);
      float4 w1 = *(const float4*)(arow + kA0 + 4);
      a[0] = (_Float16)w0.x; a[1] = (_Float16)w0.y;
      a[2] = (_Float16)w0.z; a[3] = (_Float16)w0.w;
      a[4] = (_Float16)w1.x; a[5] = (_Float16)w1.y;
      a[6] = (_Float16)w1.z; a[7] = (_Float16)w1.w;
    }
    if (kc < 64) {   // compile-time after unroll: halves 8..15 valid
      float4 w2 = *(const float4*)(arow + kA0 + 16);
      float4 w3 = *(const float4*)(arow + kA0 + 20);
      a[8]  = (_Float16)w2.x; a[9]  = (_Float16)w2.y;
      a[10] = (_Float16)w2.z; a[11] = (_Float16)w2.w;
      a[12] = (_Float16)w3.x; a[13] = (_Float16)w3.y;
      a[14] = (_Float16)w3.z; a[15] = (_Float16)w3.w;
    } else {         // K 80..95 -> zero pad
#pragma unroll
      for (int h = 8; h < 16; ++h) a[h] = (_Float16)0.f;
    }

    // ---- B fragment: 32x16 f16. Lane holds column N = lane&15 (N>=8 zeroed;
    // only 8 frames), halves = K in the same order as A (kA0+0..7, kA0+16..23).
    const int ncol   = lane & 15;
    const float bs   = (ncol < 8) ? 1.f : 0.f;
    const float* __restrict__ bcol = y + (ncol & 7);  // in-bounds
    v16h b;
#pragma unroll
    for (int h = 0; h < 8; ++h) b[h] = (_Float16)(bs * bcol[(kA0 + h) * 8]);
#pragma unroll
    for (int h = 8; h < 16; ++h)
      b[h] = (kc < 64) ? (_Float16)(bs * bcol[(kA0 + 8 + h) * 8]) : (_Float16)0.f;

    acc = __builtin_amdgcn_wmma_f32_16x16x32_f16(
        /*neg_a=*/false, a, /*neg_b=*/false, b,
        /*c_mod=*/(short)0, acc, /*reuse_a=*/false, /*reuse_b=*/false);
  }

  // D layout: VGPR r: lanes 0-15 -> (M=m0+r, N=lane); lanes 16-31 -> (M=m0+8+r).
  const int n = lane & 15;
  if (n < 8) {
    const int mb = m0 + 8 * hiK;
#pragma unroll
    for (int r = 0; r < 8; ++r) condf[n * WN_CROWS + mb + r] = acc[r];
  }
}

// ---------------------------------------------------------------------------
// Kernel 2: persistent single-workgroup autoregressive scan.
// ---------------------------------------------------------------------------
__global__ void __launch_bounds__(256, 1)
wavenet_scan(const float* __restrict__ samples,  // (512,)
             const int*   __restrict__ cptr,     // scalar
             const float* __restrict__ emb,      // (256, 64)
             const float* __restrict__ WVp,      // (10, 128, 64)
             const float* __restrict__ WVq,      // (10, 128, 64)
             const float* __restrict__ Wo,       // (9, 320, 64)
             const float* __restrict__ bo,       // (9, 320)
             const float* __restrict__ Wol,      // (256, 64)
             const float* __restrict__ bol,      // (256,)
             const float* __restrict__ e1w,      // (256, 256)
             const float* __restrict__ e1b,      // (256,)
             const float* __restrict__ e2w,      // (256, 256)
             const float* __restrict__ e2b,      // (256,)
             const float* __restrict__ condf,    // (8, 1280) frame-major
             int* __restrict__ outs) {           // (512,) int32
  extern __shared__ float sm[];
  float* ring = sm;              // per-layer ring buffers, base (2^j-1)*64
  float* xL   = sm + X_OFF;
  float* vL   = sm + V_OFF;
  float* hL   = sm + H_OFF;
  float* zL   = sm + Z_OFF;
  float* skL  = sm + SK_OFF;
  float* hidL = sm + HID_OFF;
  float* scn  = sm + SCN_OFF;
  float* red  = sm + RED_OFF;
  int*   misc = (int*)(sm + LDS_F);  // [0]=prev sample, [1]=chosen index

  const int tid  = threadIdx.x;     // 0..255
  const int lane = tid & 31;
  const int wave = tid >> 5;        // 0..7 (one 16-row gate tile per wave)

  // zero the dilation history (reference starts hist at zeros)
  for (int i = tid; i < RING_F; i += 256) ring[i] = 0.f;
  if (tid == 0) misc[0] = WN_NCLS / 2 - 1;  // prev = 127
  __syncthreads();

  const float cf = (float)cptr[0];

  for (int t = 0; t < WN_T; ++t) {
    const int prev = misc[0];
    if (tid < 64) xL[tid] = emb[prev * WN_R + tid];
    float skipAcc = 0.f;  // this thread owns skip[tid]
    __syncthreads();

    const float* crow = condf + (t >> 6) * WN_CROWS;  // frame = t / HOP

    for (int j = 0; j < WN_L; ++j) {
      const int dil = 1 << j;
      float* slot = ring + (((dil - 1) << 6)) + ((t & (dil - 1)) << 6);

      // v = concat(past, x); past comes from ring slot (read BEFORE overwrite)
      if (tid < 64)        vL[tid] = slot[tid];
      else if (tid < 128)  vL[tid] = xL[tid - 64];
      __syncthreads();

      // written[j] = x (layer input) into the same ring slot
      if (tid < 64) slot[tid] = xL[tid];

      // ---- gate matvec: h = [WV_past | WV_pres] @ v  (128x128 @ 128) ----
      const float* Wp = WVp + j * WN_G * WN_R;
      const float* Wq = WVq + j * WN_G * WN_R;
#if HAVE_WMMA_F32X4
      {
        // wave w owns rows [w*16, w*16+16). A frag (16x4 f32): lane<16 holds
        // K pair {kc,kc+1}, lane>=16 holds {kc+2,kc+3}, row M = lane&15.
        const int m    = wave * 16 + (lane & 15);
        const int ksel = (lane >> 4) << 1;  // 0 or 2
        v8f acc = {};
        for (int kc = 0; kc < 128; kc += 4) {
          const int ks = kc + ksel;
          const float* src = (ks < 64) ? (Wp + m * WN_R + ks)
                                       : (Wq + m * WN_R + (ks - 64));
          float2 w2 = *(const float2*)src;
          v2f a; a.x = w2.x; a.y = w2.y;
          // broadcast v into every B column -> all D columns equal h
          v2f b; b.x = vL[kc + ksel]; b.y = vL[kc + ksel + 1];
          acc = __builtin_amdgcn_wmma_f32_16x16x4_f32(
              /*neg_a=*/false, a, /*neg_b=*/false, b,
              /*c_mod=*/(short)0, acc, /*reuse_a=*/false, /*reuse_b=*/false);
        }
        // D: VGPR r holds M = tile+r (lanes 0-15) / tile+8+r (lanes 16-31);
        // columns identical, so lanes 0 and 16 publish the tile (+cond).
        if ((lane & 15) == 0) {
          const int mb = wave * 16 + 8 * (lane >> 4);
#pragma unroll
          for (int r = 0; r < 8; ++r)
            hL[mb + r] = acc[r] + crow[j * WN_G + mb + r];
        }
      }
#else
      // VALU fallback: threads 0..127 each produce one gate row.
      if (tid < 128) {
        float acc = crow[j * WN_G + tid];
        acc += dot64(Wp + tid * WN_R, vL);        // past half
        acc += dot64(Wq + tid * WN_R, vL + 64);   // present half
        hL[tid] = acc;
      }
#endif
      __syncthreads();

      // z = tanh(h[:64]) * sigmoid(h[64:])
      if (tid < 64) {
        const float hw = hL[tid];
        const float hf = hL[tid + 64];
        zL[tid] = tanhf(hw) * (1.f / (1.f + expf(-hf)));
      }
      __syncthreads();

      // ---- output matvec ----
      if (j < WN_L - 1) {
        const float* W = Wo + j * 320 * WN_R;
        const float* b = bo + j * 320;
        // skip contribution: o[64 + tid]
        skipAcc += dot64(W + (64 + tid) * WN_R, zL) + b[64 + tid];
        // residual update: x += o[:64]
        if (tid < 64) {
          const float o0 = dot64(W + tid * WN_R, zL) + b[tid];
          xL[tid] = xL[tid] + o0;  // own element; consumed after barrier
        }
      } else {
        skipAcc += dot64(Wol + tid * WN_R, zL) + bol[tid];
      }
      __syncthreads();
    }

    // ---- head: hid = relu(e1w @ relu(skip) + e1b); logits = (e2w@hid+e2b)*c
    skL[tid] = fmaxf(skipAcc, 0.f);
    __syncthreads();
    hidL[tid] = fmaxf(dot256(e1w + tid * 256, skL) + e1b[tid], 0.f);
    __syncthreads();
    const float lg = (dot256(e2w + tid * 256, hidL) + e2b[tid]) * cf;

    // ---- sample: argmax(cumsum(softmax(logits)) > s) ----
    red[tid] = lg;
    if (tid == 0) misc[1] = 0;  // default (argmax of all-False = 0)
    __syncthreads();
#pragma unroll
    for (int off = 128; off > 0; off >>= 1) {
      if (tid < off) red[tid] = fmaxf(red[tid], red[tid + off]);
      __syncthreads();
    }
    const float mx = red[0];
    __syncthreads();
    scn[tid] = expf(lg - mx);
    __syncthreads();
#pragma unroll
    for (int off = 1; off < 256; off <<= 1) {   // inclusive prefix sum
      const float v = (tid >= off) ? scn[tid - off] : 0.f;
      __syncthreads();
      scn[tid] += v;
      __syncthreads();
    }
    const float total = scn[255];
    const float sT = samples[t] * total;        // prefix/total > s
    const float incl = scn[tid];
    const float prevIncl = (tid == 0) ? -1.f : scn[tid - 1];
    if (incl > sT && prevIncl <= sT) misc[1] = tid;  // unique monotone boundary
    __syncthreads();
    const int newIdx = misc[1];
    if (tid == 0) {
      outs[t] = newIdx;
      misc[0] = newIdx;
    }
    __syncthreads();
  }
}

// ---------------------------------------------------------------------------
extern "C" void kernel_launch(void* const* d_in, const int* in_sizes, int n_in,
                              void* d_out, int out_size, void* d_ws,
                              size_t ws_size, hipStream_t stream) {
  const float* y       = (const float*)d_in[0];   // (80, 8)
  const float* samples = (const float*)d_in[1];   // (512,)
  const int*   c       = (const int*)d_in[2];     // scalar
  const float* emb     = (const float*)d_in[3];   // (256, 64)
  const float* condW   = (const float*)d_in[4];   // (1280, 80)
  const float* WVp     = (const float*)d_in[5];   // (10, 128, 64, 1)
  const float* WVq     = (const float*)d_in[6];   // (10, 128, 64)
  const float* Wo      = (const float*)d_in[7];   // (9, 320, 64)
  const float* bo      = (const float*)d_in[8];   // (9, 320)
  const float* Wol     = (const float*)d_in[9];   // (256, 64)
  const float* bol     = (const float*)d_in[10];  // (256,)
  const float* e1w     = (const float*)d_in[11];  // (256, 256)
  const float* e1b     = (const float*)d_in[12];  // (256,)
  const float* e2w     = (const float*)d_in[13];  // (256, 256)
  const float* e2b     = (const float*)d_in[14];  // (256,)

  float* condf = (float*)d_ws;   // (8, 1280) = 40 KB scratch
  int*   outs  = (int*)d_out;    // (512,) int32

  // Stage 1: condition GEMM (WMMA f16 -> f32 accum), 80 tiles x 1 wave.
  cond_gemm<<<80, 32, 0, stream>>>(condW, y, condf);

  // Stage 2: persistent single-WGP scan; ~267 KB dynamic LDS (ring + scratch)
  // — legal on CDNA5 (320 KB LDS per workgroup).
  wavenet_scan<<<1, 256, LDS_BYTES, stream>>>(
      samples, c, emb, WVp, WVq, Wo, bo, Wol, bol, e1w, e1b, e2w, e2b, condf,
      outs);
}
